// TransformerEncoderLayer_54769422958689
// MI455X (gfx1250) — matrix operand order repaired and checked by reading, not compiled, not run on hardware
//
#include <hip/hip_runtime.h>

#define BB 8
#define SS 2048
#define DD 1024
#define DKK 32

#define USE_ASYNC_LDS 1

typedef __bf16 bf16;
typedef __attribute__((ext_vector_type(16))) __bf16 v16bf;
typedef __attribute__((ext_vector_type(8)))  float  v8f;

#define WMMA_BF16(a, b, c) \
  __builtin_amdgcn_wmma_f32_16x16x32_bf16(false, (a), false, (b), (short)0, (c), false, false)

// ---------------------------------------------------------------------------
// Async global->LDS copy: per-lane 16B DMA, tracked by ASYNCcnt (CDNA5
// GLOBAL_LOAD_ASYNC_TO_LDS_B128). Generic LDS pointer low 32 bits are the LDS
// byte offset (ISA 10.2 aperture rules).
__device__ __forceinline__ void async_b128(void* lds, const void* g) {
#if USE_ASYNC_LDS
  asm volatile("global_load_async_to_lds_b128 %0, %1, off"
               :: "v"((unsigned)(unsigned long long)lds), "v"(g)
               : "memory");
#else
  *(uint4*)lds = *(const uint4*)g;
#endif
}
__device__ __forceinline__ void async_wait() {
#if USE_ASYNC_LDS
  asm volatile("s_wait_asynccnt 0x0" ::: "memory");
#endif
}

// ---------------------------------------------------------------------------
// Fragment loaders (contiguous k order, k = 16*(lane/16) + s, same for A and B; not the ISA table order)
// A 16x32 bf16: lanes 0-15 row M=lane, K=0..15; lanes 16-31 row
// M=lane-16, K=16..31.
__device__ __forceinline__ v16bf frag_a(const bf16* tile, int ld, int m0) {
  const int lane = threadIdx.x & 31;
  const int row  = m0 + (lane & 15);
  const int kb   = (lane < 16) ? 0 : 16;
  const bf16* p  = tile + row * ld;
  v16bf a;
#pragma unroll
  for (int j = 0; j < 8; ++j) {
    const int k = kb + 2 * j;
    a[2 * j]     = p[k];
    a[2 * j + 1] = p[k + 1];
  }
  return a;
}

// B 32x16 bf16 stored [N][K] in LDS: lanes 0-15 col N=lane K=0..15; lanes
// 16-31 col N=lane-16 K=16..31.
__device__ __forceinline__ v16bf frag_b(const bf16* tile, int ld, int n0) {
  const int lane = threadIdx.x & 31;
  const int col  = n0 + (lane & 15);
  const int kb   = (lane < 16) ? 0 : 16;
  const bf16* p  = tile + col * ld + kb;
  v16bf b;
#pragma unroll
  for (int j = 0; j < 16; ++j) b[j] = p[j];
  return b;
}

// ---------------------------------------------------------------------------
__global__ void zero_f32(float* __restrict__ p, int n) {
  int i = blockIdx.x * 256 + threadIdx.x;
  if (i < n) p[i] = 0.0f;
}

// ---------------------------------------------------------------------------
// Q/K projection: [BS,1024]x[1024,32], 128x32 tile, fused row L2-norm.
__global__ __launch_bounds__(256) void qk_proj_kernel(
    const float* __restrict__ x, const float* __restrict__ Wq,
    const float* __restrict__ Wk, bf16* __restrict__ qn, bf16* __restrict__ kn) {
  __shared__ bf16  lA[128][34];
  __shared__ bf16  lB[32][34];
  __shared__ float lQ[128][33];
  const int t   = threadIdx.x;
  const int w   = t >> 5;
  const int m0  = w * 16;
  const int bs0 = blockIdx.x * 128;
  const float* W = blockIdx.y ? Wk : Wq;
  bf16* out      = blockIdx.y ? kn : qn;

  v8f acc0 = {}, acc1 = {};
  for (int kt = 0; kt < DD / 32; ++kt) {
    const int k0 = kt * 32;
#pragma unroll
    for (int i = 0; i < 4; ++i) {             // x tile 128x32, float4 loads
      int idx = i * 256 + t;
      int r = idx >> 3, c4 = (idx & 7) * 4;
      float4 f = *(const float4*)&x[(size_t)(bs0 + r) * DD + k0 + c4];
      lA[r][c4 + 0] = (bf16)f.x; lA[r][c4 + 1] = (bf16)f.y;
      lA[r][c4 + 2] = (bf16)f.z; lA[r][c4 + 3] = (bf16)f.w;
    }
    {                                         // W tile 32x32 transposed
      int kk = t >> 3, n4 = (t & 7) * 4;
      float4 f = *(const float4*)&W[(size_t)(k0 + kk) * DKK + n4];
      lB[n4 + 0][kk] = (bf16)f.x; lB[n4 + 1][kk] = (bf16)f.y;
      lB[n4 + 2][kk] = (bf16)f.z; lB[n4 + 3][kk] = (bf16)f.w;
    }
    if (kt + 1 < DD / 32)
      __builtin_prefetch(&x[(size_t)(bs0 + (t >> 1)) * DD + k0 + 32], 0, 0);
    __syncthreads();
    v16bf a  = frag_a(&lA[0][0], 34, m0);
    v16bf b0 = frag_b(&lB[0][0], 34, 0);
    v16bf b1 = frag_b(&lB[0][0], 34, 16);
    acc0 = WMMA_BF16(a, b0, acc0);
    acc1 = WMMA_BF16(a, b1, acc1);
    __syncthreads();
  }
  const int lane = t & 31;
  const int nn   = lane & 15;
  const int mb   = (lane < 16) ? 0 : 8;
#pragma unroll
  for (int j = 0; j < 8; ++j) {
    lQ[m0 + mb + j][nn]      = acc0[j];
    lQ[m0 + mb + j][16 + nn] = acc1[j];
  }
  __syncthreads();
  if (t < 128) {                              // fused L2 normalize (DK=32)
    float ss = 0.0f;
#pragma unroll
    for (int c = 0; c < 32; ++c) { float v = lQ[t][c]; ss += v * v; }
    const float scale = 1.0f / fmaxf(sqrtf(ss), 1e-12f);
#pragma unroll
    for (int c = 0; c < 32; ++c)
      out[(size_t)(bs0 + t) * DKK + c] = (bf16)(lQ[t][c] * scale);
  }
}

// ---------------------------------------------------------------------------
// Generic [BS,1024]x[1024,1024] GEMM, 256x128 tile, 16 WMMA/wave/K-step.
// mode 0: A=f32 x, store bf16 (V projection).
// mode 1: A=bf16 zz (async-LDS path), +bias, LeakyReLU, +residual, store f32.
__global__ __launch_bounds__(256) void gemm_dxd_kernel(
    const float* __restrict__ A32, const bf16* __restrict__ A16,
    const float* __restrict__ W, const float* __restrict__ bias,
    const float* __restrict__ res, bf16* __restrict__ out16,
    float* __restrict__ out32, int mode) {
  __shared__ bf16 lA[256][40];   // 80B pitch: b128-aligned rows for async DMA
  __shared__ bf16 lB[128][34];
  const int t   = threadIdx.x;
  const int w   = t >> 5;
  const int m0  = w * 32;
  const int bs0 = blockIdx.x * 256;
  const int n0g = blockIdx.y * 128;

  v8f acc[2][8] = {};
  for (int kt = 0; kt < DD / 32; ++kt) {
    const int k0 = kt * 32;
    if (mode == 0) {                          // f32 A tile -> bf16, float4
#pragma unroll
      for (int i = 0; i < 8; ++i) {
        int idx = i * 256 + t;
        int r = idx >> 3, c4 = (idx & 7) * 4;
        float4 f = *(const float4*)&A32[(size_t)(bs0 + r) * DD + k0 + c4];
        lA[r][c4 + 0] = (bf16)f.x; lA[r][c4 + 1] = (bf16)f.y;
        lA[r][c4 + 2] = (bf16)f.z; lA[r][c4 + 3] = (bf16)f.w;
      }
    } else {                                  // bf16 A tile via async DMA
#pragma unroll
      for (int i = 0; i < 4; ++i) {
        int idx = i * 256 + t;
        int r = idx >> 2, c8 = (idx & 3) * 8;
        async_b128(&lA[r][c8], &A16[(size_t)(bs0 + r) * DD + k0 + c8]);
      }
    }
#pragma unroll
    for (int i = 0; i < 4; ++i) {             // W tile 32x128 transposed
      int idx = i * 256 + t;
      int kk = idx >> 5, n4 = (idx & 31) * 4;
      float4 f = *(const float4*)&W[(size_t)(k0 + kk) * DD + n0g + n4];
      lB[n4 + 0][kk] = (bf16)f.x; lB[n4 + 1][kk] = (bf16)f.y;
      lB[n4 + 2][kk] = (bf16)f.z; lB[n4 + 3][kk] = (bf16)f.w;
    }
    if (kt + 1 < DD / 32)
      __builtin_prefetch(&W[(size_t)(k0 + 32 + (t >> 7)) * DD + n0g + (t & 127)], 0, 0);
    if (mode != 0) async_wait();
    __syncthreads();
    v16bf a0 = frag_a(&lA[0][0], 40, m0);
    v16bf a1 = frag_a(&lA[0][0], 40, m0 + 16);
#pragma unroll
    for (int ns = 0; ns < 8; ++ns) {
      v16bf b = frag_b(&lB[0][0], 34, ns * 16);
      acc[0][ns] = WMMA_BF16(a0, b, acc[0][ns]);
      acc[1][ns] = WMMA_BF16(a1, b, acc[1][ns]);
    }
    __syncthreads();
  }
  const int lane = t & 31;
  const int nn   = lane & 15;
  const int mb   = (lane < 16) ? 0 : 8;
#pragma unroll
  for (int ms = 0; ms < 2; ++ms) {
#pragma unroll
    for (int ns = 0; ns < 8; ++ns) {
#pragma unroll
      for (int j = 0; j < 8; ++j) {
        const int row = bs0 + m0 + ms * 16 + mb + j;
        const int col = n0g + ns * 16 + nn;
        float v = acc[ms][ns][j];
        if (mode == 0) {
          out16[(size_t)row * DD + col] = (bf16)v;
        } else {
          v += bias[col];
          v = (v >= 0.0f) ? v : 0.01f * v;    // LeakyReLU
          out32[(size_t)row * DD + col] = res[(size_t)row * DD + col] + v;
        }
      }
    }
  }
}

// ---------------------------------------------------------------------------
// Scores: 64x64 causal tile, z = exp(mask(q k^T / sqrt(32))) bf16, row-sums
// via LDS + f32 global atomics. Upper-triangle blocks exit early (uniform
// branch keeps EXEC all-ones for WMMA).
__global__ __launch_bounds__(256) void scores_kernel(
    const bf16* __restrict__ qn, const bf16* __restrict__ kn,
    bf16* __restrict__ z, float* __restrict__ rowsum) {
  const int jt = blockIdx.x, it = blockIdx.y, b = blockIdx.z;
  if (jt > it) return;
  __shared__ bf16  lQ[64][40];
  __shared__ bf16  lK[64][40];
  __shared__ float sT[64][65];
  const int t  = threadIdx.x;
  const int q0 = it * 64, s0 = jt * 64;
  {
    int r = t >> 2, c8 = (t & 3) * 8;         // async DMA both tiles
    async_b128(&lQ[r][c8], &qn[(size_t)(b * SS + q0 + r) * DKK + c8]);
    async_b128(&lK[r][c8], &kn[(size_t)(b * SS + s0 + r) * DKK + c8]);
  }
  async_wait();
  __syncthreads();
  const int w  = t >> 5;
  const int m0 = (w & 3) * 16;
  const int n0 = (w >> 2) * 32;
  v16bf a  = frag_a(&lQ[0][0], 40, m0);
  v16bf b0 = frag_b(&lK[0][0], 40, n0);       // k tile is naturally [N][K]
  v16bf b1 = frag_b(&lK[0][0], 40, n0 + 16);
  v8f acc0 = {}, acc1 = {};
  acc0 = WMMA_BF16(a, b0, acc0);              // one step: K = DK = 32
  acc1 = WMMA_BF16(a, b1, acc1);
  const float rscale = 0.17677669529663687f;  // 1/sqrt(32)
  const int lane = t & 31;
  const int nn   = lane & 15;
  const int mb   = (lane < 16) ? 0 : 8;
#pragma unroll
  for (int j = 0; j < 8; ++j) {
    const int row = m0 + mb + j;
    const int qg  = q0 + row;
    const int c0 = n0 + nn, c1 = n0 + 16 + nn;
    float e0 = (s0 + c0 > qg) ? 0.0f : __expf(acc0[j] * rscale);
    float e1 = (s0 + c1 > qg) ? 0.0f : __expf(acc1[j] * rscale);
    sT[row][c0] = e0; sT[row][c1] = e1;
    size_t zrow = ((size_t)b * SS + qg) * SS + s0;
    z[zrow + c0] = (bf16)e0;
    z[zrow + c1] = (bf16)e1;
  }
  __syncthreads();
  if (t < 64) {
    float s = 0.0f;
#pragma unroll
    for (int c = 0; c < 64; ++c) s += sT[t][c];
    atomicAdd(&rowsum[b * SS + q0 + t], s);
  }
}

// ---------------------------------------------------------------------------
// PV: sa = (z @ V) * 1/(rowsum+1), 64x128 tile, K bounded by causal wedge.
__global__ __launch_bounds__(256) void pv_kernel(
    const bf16* __restrict__ z, const bf16* __restrict__ vb,
    const float* __restrict__ rowsum, float* __restrict__ sa) {
  __shared__ bf16 lA[64][40];
  __shared__ bf16 lB[128][34];
  const int t   = threadIdx.x;
  const int n0g = blockIdx.x * 128;
  const int it  = blockIdx.y;
  const int b   = blockIdx.z;
  const int q0  = it * 64;
  const int w   = t >> 5;
  const int m0  = (w & 3) * 16;
  const int nw  = (w >> 2) * 64;
  v8f acc[4] = {};
  const int nkt = (it + 1) * 2;               // only s <= q tiles contribute
  for (int kt = 0; kt < nkt; ++kt) {
    const int s0 = kt * 32;
    {                                         // z tile 64x32 via async DMA
      int r = t >> 2, c8 = (t & 3) * 8;
      async_b128(&lA[r][c8], &z[((size_t)b * SS + q0 + r) * SS + s0 + c8]);
    }
#pragma unroll
    for (int i = 0; i < 2; ++i) {             // V tile 32x128 transposed
      int idx = i * 256 + t;
      int kk = idx >> 4, n8 = (idx & 15) * 8;
      uint4 u = *(const uint4*)&vb[(size_t)(b * SS + s0 + kk) * DD + n0g + n8];
      const bf16* e = (const bf16*)&u;
#pragma unroll
      for (int q = 0; q < 8; ++q) lB[n8 + q][kk] = e[q];
    }
    if (kt + 1 < nkt)
      __builtin_prefetch(&vb[(size_t)(b * SS + s0 + 32 + (t >> 7)) * DD + n0g + (t & 127)], 0, 0);
    async_wait();
    __syncthreads();
    v16bf a = frag_a(&lA[0][0], 40, m0);
#pragma unroll
    for (int ns = 0; ns < 4; ++ns) {
      v16bf bb = frag_b(&lB[0][0], 34, nw + ns * 16);
      acc[ns] = WMMA_BF16(a, bb, acc[ns]);
    }
    __syncthreads();
  }
  const int lane = t & 31;
  const int nn   = lane & 15;
  const int mb   = (lane < 16) ? 0 : 8;
#pragma unroll
  for (int j = 0; j < 8; ++j) {
    const int qg   = q0 + m0 + mb + j;
    const float sc = 1.0f / (rowsum[b * SS + qg] + 1.0f);
#pragma unroll
    for (int ns = 0; ns < 4; ++ns)
      sa[((size_t)b * SS + qg) * DD + n0g + nw + ns * 16 + nn] = acc[ns][j] * sc;
  }
}

// ---------------------------------------------------------------------------
// LayerNorm over D=1024; optional residual input; optional bf16 copy out.
__global__ __launch_bounds__(256) void ln_kernel(
    const float* __restrict__ a, const float* __restrict__ bsrc,
    const float* __restrict__ g, const float* __restrict__ beta,
    float* __restrict__ out32, bf16* __restrict__ out16) {
  __shared__ float redA[8];
  __shared__ float redB[8];
  __shared__ float bc[2];
  const int row     = blockIdx.x;
  const size_t base = (size_t)row * DD;
  const int t = threadIdx.x;
  const int c0 = t * 4;
  float4 v = *(const float4*)&a[base + c0];
  if (bsrc) {
    float4 r2 = *(const float4*)&bsrc[base + c0];
    v.x += r2.x; v.y += r2.y; v.z += r2.z; v.w += r2.w;
  }
  float s  = v.x + v.y + v.z + v.w;
  float s2 = v.x * v.x + v.y * v.y + v.z * v.z + v.w * v.w;
  for (int off = 16; off > 0; off >>= 1) {
    s  += __shfl_xor(s, off, 32);
    s2 += __shfl_xor(s2, off, 32);
  }
  if ((t & 31) == 0) { redA[t >> 5] = s; redB[t >> 5] = s2; }
  __syncthreads();
  if (t == 0) {
    float ts = 0.0f, ts2 = 0.0f;
#pragma unroll
    for (int i = 0; i < 8; ++i) { ts += redA[i]; ts2 += redB[i]; }
    float mu  = ts / (float)DD;
    float var = ts2 / (float)DD - mu * mu;
    bc[0] = mu; bc[1] = rsqrtf(var + 1e-5f);
  }
  __syncthreads();
  const float mu = bc[0], rs = bc[1];
  float4 gv = *(const float4*)&g[c0];
  float4 bv = *(const float4*)&beta[c0];
  float y0 = (v.x - mu) * rs * gv.x + bv.x;
  float y1 = (v.y - mu) * rs * gv.y + bv.y;
  float y2 = (v.z - mu) * rs * gv.z + bv.z;
  float y3 = (v.w - mu) * rs * gv.w + bv.w;
  float4 yo = {y0, y1, y2, y3};
  *(float4*)&out32[base + c0] = yo;
  if (out16) {
    out16[base + c0 + 0] = (bf16)y0; out16[base + c0 + 1] = (bf16)y1;
    out16[base + c0 + 2] = (bf16)y2; out16[base + c0 + 3] = (bf16)y3;
  }
}

// ---------------------------------------------------------------------------
extern "C" void kernel_launch(void* const* d_in, const int* in_sizes, int n_in,
                              void* d_out, int out_size, void* d_ws, size_t ws_size,
                              hipStream_t stream) {
  (void)in_sizes; (void)n_in; (void)out_size; (void)ws_size;
  const float* x  = (const float*)d_in[0];
  // d_in[1] = attn_mask (causal; recomputed analytically, not read)
  const float* Wq = (const float*)d_in[2];
  const float* Wk = (const float*)d_in[3];
  const float* Wv = (const float*)d_in[4];
  const float* Wf = (const float*)d_in[5];
  const float* bf = (const float*)d_in[6];
  const float* g1 = (const float*)d_in[7];
  const float* b1 = (const float*)d_in[8];
  const float* g2 = (const float*)d_in[9];
  const float* b2 = (const float*)d_in[10];
  float* out = (float*)d_out;

  char* ws = (char*)d_ws;
  size_t off = 0;
  auto carve = [&](size_t bytes) -> void* {
    void* p = ws + off;
    off += (bytes + 255) & ~(size_t)255;
    return p;
  };
  const size_t BS = (size_t)BB * SS;                 // 16384
  bf16*  qn     = (bf16*)carve(BS * DKK * 2);        //  1 MB
  bf16*  kn     = (bf16*)carve(BS * DKK * 2);        //  1 MB
  bf16*  vb     = (bf16*)carve(BS * DD * 2);         // 32 MB (reused as zz bf16)
  bf16*  zm     = (bf16*)carve((size_t)BB * SS * SS * 2); // 64 MB
  float* rowsum = (float*)carve(BS * 4);             // 64 KB
  float* sa     = (float*)carve(BS * DD * 4);        // 64 MB (reused as t)
  float* zz     = (float*)carve(BS * DD * 4);        // 64 MB
  bf16*  zzb  = vb;   // V dead after pv_kernel
  float* tbuf = sa;   // sa dead after ln1

  zero_f32<<<dim3((unsigned)((BS + 255) / 256)), dim3(256), 0, stream>>>(rowsum, (int)BS);
  qk_proj_kernel<<<dim3((unsigned)(BS / 128), 2), dim3(256), 0, stream>>>(x, Wq, Wk, qn, kn);
  gemm_dxd_kernel<<<dim3((unsigned)(BS / 256), DD / 128), dim3(256), 0, stream>>>(
      x, nullptr, Wv, nullptr, nullptr, vb, nullptr, 0);
  scores_kernel<<<dim3(SS / 64, SS / 64, BB), dim3(256), 0, stream>>>(qn, kn, zm, rowsum);
  pv_kernel<<<dim3(DD / 128, SS / 64, BB), dim3(256), 0, stream>>>(zm, vb, rowsum, sa);
  ln_kernel<<<dim3((unsigned)BS), dim3(256), 0, stream>>>(x, sa, g1, b1, zz, zzb);
  gemm_dxd_kernel<<<dim3((unsigned)(BS / 256), DD / 128), dim3(256), 0, stream>>>(
      nullptr, zzb, Wf, bf, zz, nullptr, tbuf, 1);
  ln_kernel<<<dim3((unsigned)BS), dim3(256), 0, stream>>>(tbuf, nullptr, g2, b2, out, nullptr);
}
